// LSHmodule_26740466385571
// MI455X (gfx1250) — compile-verified, matrix operand order checked by reading, not run on hardware
//
#include <hip/hip_runtime.h>
#include <hip/hip_bf16.h>

typedef __bf16 bf16_t;
typedef __attribute__((ext_vector_type(16))) __bf16 v16bf;
typedef __attribute__((ext_vector_type(8)))  __bf16 bf16x8;
typedef __attribute__((ext_vector_type(4)))  __bf16 bf16x4;
typedef __attribute__((ext_vector_type(8)))  float  v8f;
typedef __attribute__((ext_vector_type(4)))  unsigned int v4u;
typedef __attribute__((ext_vector_type(4)))  int v4i;

static constexpr int Sg = 2048;   // sequence length
static constexpr int Eg = 1024;   // embed dim
static constexpr int Hh = 16;     // heads
static constexpr int Dd = 64;     // head dim

// ---------------- gfx1250 async global->LDS copy (guarded) ----------------
#if defined(__HIP_DEVICE_COMPILE__) && __has_builtin(__builtin_amdgcn_global_load_async_to_lds_b128)
#define ASYNC_LDS 1
#else
#define ASYNC_LDS 0
#endif

__device__ __forceinline__ void async_cp16(const void* g, void* l) {
#if ASYNC_LDS
  __builtin_amdgcn_global_load_async_to_lds_b128(
      (__attribute__((address_space(1))) v4i*)g,
      (__attribute__((address_space(3))) v4i*)l, 0, 0);
#else
  *(v4u*)l = *(const v4u*)g;
#endif
}

__device__ __forceinline__ void async_wait0() {
#if ASYNC_LDS
#if __has_builtin(__builtin_amdgcn_s_wait_asynccnt)
  __builtin_amdgcn_s_wait_asynccnt(0);
#else
  asm volatile("s_wait_asynccnt 0" ::: "memory");
#endif
#endif
}

// ---- WMMA fragment helpers (bf16 16x16x32, wave32) -----------------------
// A (16x32 MxK): lane l holds row M=l%16; VGPR i pair k=(i<4?2i:16+2(i-4))+8*(l/16)
//   => VGPRs 0-3 = k[8h..8h+7], VGPRs 4-7 = k[16+8h..16+8h+7]: two b128 loads.
// B (32x16 KxN): lane l holds col N=l%16 with the same K mapping, so storing
//   the B tile [n][k] row-major makes B fragments identical two-b128 loads.
// C (16x16): lane l holds col N=l%16, row M = vgpr + 8*(l/16).

__device__ __forceinline__ v16bf load_frag_row_bf16(const bf16_t* rowbase, int half) {
  union { v16bf v; v4u q[2]; } u;
  u.q[0] = *(const v4u*)(rowbase + 8 * half);
  u.q[1] = *(const v4u*)(rowbase + 16 + 8 * half);
  return u.v;
}

// same fragment, sourced from a raw f32 row (convert at load: v_cvt_pk_bf16_f32)
__device__ __forceinline__ v16bf load_frag_row_f32(const float* rowbase, int half) {
  v16bf v;
#pragma unroll
  for (int i = 0; i < 8; ++i) {
    const int k = ((i < 4) ? (2 * i) : (16 + 2 * (i - 4))) + 8 * half;
    v[2 * i]     = (bf16_t)rowbase[k];
    v[2 * i + 1] = (bf16_t)rowbase[k + 1];
  }
  return v;
}

__device__ __forceinline__ v8f wmma_bf16(v16bf a, v16bf b, v8f c) {
  return __builtin_amdgcn_wmma_f32_16x16x32_bf16(
      false, a, false, b, (short)0, c, false, false);
}

// =========================================================================
// Kernel 1: Q/V projection GEMM.  q = x @ W^T + b,  M=4096, N=1024, K=1024.
// A tile: x rows, f32->bf16 packed b64 stores, [m][k] (stride 48 for banks).
// B tile: W rows copied RAW f32 via async-to-LDS, [n][k]; converted in the
// fragment load.  blockIdx.z selects q vs v.  Output bf16 [B,H,S,64].
// =========================================================================
__global__ __launch_bounds__(256) void proj_gemm_kernel(
    const float* __restrict__ x,
    const float* __restrict__ Wq, const float* __restrict__ bq,
    const float* __restrict__ Wv, const float* __restrict__ bv,
    bf16_t* __restrict__ qbf, bf16_t* __restrict__ vbf) {
  constexpr int TM = 128, TK = 32;
  constexpr int AP = TK + 16;   // bf16 row stride 96B (16B-aligned, conflict-spread)
  constexpr int BPf = TK + 4;   // f32 row stride 144B (16B-aligned)
  __shared__ bf16_t ldsA[TM][AP];    // [m][k] bf16
  __shared__ float  ldsB[128][BPf];  // [n][k] raw f32 (async destination)

  const float*  W    = blockIdx.z ? Wv : Wq;
  const float*  bias = blockIdx.z ? bv : bq;
  bf16_t*       outp = blockIdx.z ? vbf : qbf;

  const int tid  = threadIdx.x;
  const int lane = tid & 31;
  const int wave = tid >> 5;
  const int wm   = wave & 3;    // 4 row-waves * 32 rows
  const int wn   = wave >> 2;   // 2 col-waves * 64 cols
  const int m0   = blockIdx.x * TM;
  const int n0   = blockIdx.y * TM;
  const int ln16 = lane & 15;
  const int half = lane >> 4;

  v8f acc[2][4] = {};

  const int ar  = tid >> 3;        // 0..31 : row within 32-row pass
  const int acm = (tid & 7) * 4;   // 0..28 : k start (4 f32 = 16B)

  for (int k0 = 0; k0 < Eg; k0 += TK) {
#pragma unroll
    for (int p = 0; p < 4; ++p) {
      const int r = p * 32 + ar;
      // A: convert f32 -> bf16, packed 8B store
      const float4 fa = *(const float4*)&x[(size_t)(m0 + r) * Eg + k0 + acm];
      bf16x4 pa;
      pa[0] = (bf16_t)fa.x; pa[1] = (bf16_t)fa.y;
      pa[2] = (bf16_t)fa.z; pa[3] = (bf16_t)fa.w;
      *(bf16x4*)&ldsA[r][acm] = pa;
      // B: raw f32 row of W, async DMA into LDS
      async_cp16(&W[(size_t)(n0 + r) * Eg + k0 + acm], &ldsB[r][acm]);
    }
    async_wait0();
    __syncthreads();

    const v16bf af0 = load_frag_row_bf16(&ldsA[wm * 32 + ln16][0], half);
    const v16bf af1 = load_frag_row_bf16(&ldsA[wm * 32 + 16 + ln16][0], half);
#pragma unroll
    for (int ct = 0; ct < 4; ++ct) {
      const v16bf bf = load_frag_row_f32(&ldsB[wn * 64 + ct * 16 + ln16][0], half);
      acc[0][ct] = wmma_bf16(af0, bf, acc[0][ct]);
      acc[1][ct] = wmma_bf16(af1, bf, acc[1][ct]);
    }
    __syncthreads();
  }

  // epilogue: bias add, convert, scatter to [B,H,S,64]
#pragma unroll
  for (int ct = 0; ct < 4; ++ct) {
    const int n  = n0 + wn * 64 + ct * 16 + ln16;
    const float bb = bias[n];
    const int h = n >> 6;
    const int d = n & 63;
#pragma unroll
    for (int s = 0; s < 2; ++s) {
#pragma unroll
      for (int r = 0; r < 8; ++r) {
        const int m    = m0 + wm * 32 + s * 16 + r + 8 * half;
        const int bidx = m >> 11;
        const int srow = m & 2047;
        const size_t o = (((size_t)bidx * Hh + h) * Sg + srow) * Dd + d;
        outp[o] = (bf16_t)(acc[s][ct][r] + bb);
      }
    }
  }
}

// =========================================================================
// Kernel 2: LSH buckets.  One thread per (b,h,s).
// =========================================================================
__global__ __launch_bounds__(256) void lsh_buckets_kernel(
    const bf16_t* __restrict__ qbf, const float* __restrict__ hp,
    int* __restrict__ bkt) {
  const int idx = blockIdx.x * 256 + threadIdx.x;
  const bf16_t* qp = qbf + (size_t)idx * Dd;
  float proj[6];
#pragma unroll
  for (int j = 0; j < 6; ++j) proj[j] = hp[64 * 6 + j];  // ones column
  for (int d = 0; d < Dd; ++d) {
    const float qv = (float)qp[d];
#pragma unroll
    for (int j = 0; j < 6; ++j) proj[j] = fmaf(qv, hp[d * 6 + j], proj[j]);
  }
  int code = 0;
#pragma unroll
  for (int j = 0; j < 6; ++j)
    if (proj[j] >= 0.0f) code |= (1 << j);
  bkt[idx] = code;
}

// =========================================================================
// Kernel 3: flash-style LSH attention, double-buffered async staging.
// grid = (S/128, B*H); 8 waves; wave owns 16 query rows.
// K tile ([key][d], B-operand of q.k^T) is an async B128 DMA; V^T tile
// ([d][key], B-operand of P.V) is a manual transpose; stage(t+1) overlaps
// with compute(t).
// =========================================================================
__global__ __launch_bounds__(256) void lsh_attn_kernel(
    const bf16_t* __restrict__ qbf, const bf16_t* __restrict__ vbf,
    const int* __restrict__ bkt, float* __restrict__ out) {
  constexpr int KT = 64;          // keys per tile
  constexpr int KP = KT + 8;      // padded row stride (144B, 16B-aligned)
  __shared__ bf16_t ldsK [2][KT][KP];    // [key][d]  (B-frag rows for scores)
  __shared__ bf16_t ldsVT[2][Dd][KP];    // [d][key]  (B-frag rows for P@V)
  __shared__ bf16_t ldsP[8][16][KT];     // per-wave P tile (A-frag rows)
  __shared__ int    ldsKB[2][KT];        // key buckets

  const int tid  = threadIdx.x;
  const int lane = tid & 31;
  const int wave = tid >> 5;
  const int ln16 = lane & 15;
  const int half = lane >> 4;

  const int bh = blockIdx.y;
  const int bb = bh >> 4;
  const int hh = bh & 15;
  const bf16_t* qh = qbf + (size_t)bh * Sg * Dd;
  const bf16_t* vh = vbf + (size_t)bh * Sg * Dd;
  const int* bktH  = bkt + (size_t)bh * Sg;
  const int q0     = blockIdx.x * 128;
  const int qrow   = q0 + wave * 16 + ln16;

  // resident query A-fragments (K = 64 -> 2 chunks of 32)
  v16bf qfrag[2];
#pragma unroll
  for (int c = 0; c < 2; ++c)
    qfrag[c] = load_frag_row_bf16(qh + (size_t)qrow * Dd + 32 * c, half);

  int qb[8];
#pragma unroll
  for (int r = 0; r < 8; ++r) qb[r] = bktH[q0 + wave * 16 + r + 8 * half];

  float rowm[8], rowl[8];
#pragma unroll
  for (int r = 0; r < 8; ++r) { rowm[r] = -1e30f; rowl[r] = 0.0f; }
  v8f oacc[4] = {};

  const int cr = tid >> 3;         // key within 32-key pass
  const int cd = (tid & 7) * 8;    // d start (8 bf16 = 16B)

  auto stage = [&](int kb0, int buf) {
#pragma unroll
    for (int p = 0; p < 2; ++p) {
      const int key = p * 32 + cr;
      // K: straight row copy -> async DMA
      async_cp16(qh + (size_t)(kb0 + key) * Dd + cd, &ldsK[buf][key][cd]);
      // V: transpose on store (d-major rows for contiguous B-fragments)
      const bf16x8 vv = *(const bf16x8*)(vh + (size_t)(kb0 + key) * Dd + cd);
#pragma unroll
      for (int j = 0; j < 8; ++j) ldsVT[buf][cd + j][key] = vv[j];
    }
    if (tid < KT) ldsKB[buf][tid] = bktH[kb0 + tid];
  };

  stage(0, 0);

  for (int t = 0; t < Sg / KT; ++t) {
    async_wait0();
    __syncthreads();
    if (t + 1 < Sg / KT) stage((t + 1) * KT, (t + 1) & 1);  // overlap next DMA
    const int buf = t & 1;

    // scores: 16 queries x 64 keys  (4 col-tiles x 2 K-chunks)
    v8f sacc[4] = {};
#pragma unroll
    for (int ct = 0; ct < 4; ++ct) {
#pragma unroll
      for (int c = 0; c < 2; ++c) {
        const v16bf kf = load_frag_row_bf16(&ldsK[buf][ct * 16 + ln16][32 * c], half);
        sacc[ct] = wmma_bf16(qfrag[c], kf, sacc[ct]);
      }
    }

    int kbv[4];
#pragma unroll
    for (int ct = 0; ct < 4; ++ct) kbv[ct] = ldsKB[buf][ct * 16 + ln16];

    // coeff + online softmax (row reductions across the 16-lane C group)
#pragma unroll
    for (int r = 0; r < 8; ++r) {
      float mx = -1e30f;
#pragma unroll
      for (int ct = 0; ct < 4; ++ct) {
        const float coeff = (qb[r] == kbv[ct]) ? 63.0f : 62.0f;
        const float z = sacc[ct][r] * (0.03125f * coeff);  // 1/sqrt(1024)
        sacc[ct][r] = z;
        mx = fmaxf(mx, z);
      }
      mx = fmaxf(mx, __shfl_xor(mx, 1, 32));
      mx = fmaxf(mx, __shfl_xor(mx, 2, 32));
      mx = fmaxf(mx, __shfl_xor(mx, 4, 32));
      mx = fmaxf(mx, __shfl_xor(mx, 8, 32));
      const float nm    = fmaxf(rowm[r], mx);
      const float alpha = __expf(rowm[r] - nm);
      rowm[r] = nm;
      float ps = 0.0f;
#pragma unroll
      for (int ct = 0; ct < 4; ++ct) {
        const float pv = __expf(sacc[ct][r] - nm);
        sacc[ct][r] = pv;
        ps += pv;
      }
      ps += __shfl_xor(ps, 1, 32);
      ps += __shfl_xor(ps, 2, 32);
      ps += __shfl_xor(ps, 4, 32);
      ps += __shfl_xor(ps, 8, 32);
      rowl[r] = rowl[r] * alpha + ps;
#pragma unroll
      for (int ot = 0; ot < 4; ++ot) oacc[ot][r] *= alpha;
    }

    // P (C-layout) -> LDS A-rows (intra-wave, DS in-order)
#pragma unroll
    for (int ct = 0; ct < 4; ++ct)
#pragma unroll
      for (int r = 0; r < 8; ++r)
        ldsP[wave][r + 8 * half][ct * 16 + ln16] = (bf16_t)sacc[ct][r];

    // O += P @ V   (K = 64 keys -> 2 chunks; 4 output col-tiles over Dh)
#pragma unroll
    for (int c = 0; c < 2; ++c) {
      const v16bf pf = load_frag_row_bf16(&ldsP[wave][ln16][32 * c], half);
#pragma unroll
      for (int ot = 0; ot < 4; ++ot) {
        const v16bf vf = load_frag_row_bf16(&ldsVT[buf][ot * 16 + ln16][32 * c], half);
        oacc[ot] = wmma_bf16(pf, vf, oacc[ot]);
      }
    }
    // no trailing barrier: next iteration's wait+barrier protects the buffers
  }

  // normalize + store to [B, S, E] with head-major E
#pragma unroll
  for (int ot = 0; ot < 4; ++ot) {
    const int d = ot * 16 + ln16;
#pragma unroll
    for (int r = 0; r < 8; ++r) {
      const int srow = q0 + wave * 16 + r + 8 * half;
      const size_t o = ((size_t)bb * Sg + srow) * Eg + hh * Dd + d;
      out[o] = oacc[ot][r] / rowl[r];
    }
  }
}

// =========================================================================
extern "C" void kernel_launch(void* const* d_in, const int* in_sizes, int n_in,
                              void* d_out, int out_size, void* d_ws, size_t ws_size,
                              hipStream_t stream) {
  const float* x  = (const float*)d_in[0];
  const float* Wq = (const float*)d_in[1];
  const float* bq = (const float*)d_in[2];
  const float* Wv = (const float*)d_in[3];
  const float* bv = (const float*)d_in[4];
  const float* hp = (const float*)d_in[5];

  char* ws = (char*)d_ws;
  bf16_t* qbf = (bf16_t*)ws;                               // 8 MB
  bf16_t* vbf = (bf16_t*)(ws + (size_t)8 * 1024 * 1024);   // 8 MB
  int*    bkt = (int*)(ws + (size_t)16 * 1024 * 1024);     // 256 KB

  dim3 g1(4096 / 128, 1024 / 128, 2);
  proj_gemm_kernel<<<g1, 256, 0, stream>>>(x, Wq, bq, Wv, bv, qbf, vbf);

  lsh_buckets_kernel<<<(2 * 16 * 2048) / 256, 256, 0, stream>>>(qbf, hp, bkt);

  dim3 g2(2048 / 128, 2 * 16);
  lsh_attn_kernel<<<g2, 256, 0, stream>>>(qbf, vbf, bkt, (float*)d_out);
}
